// vTrack_24111946400332
// MI455X (gfx1250) — matrix operand order repaired and checked by reading, not compiled, not run on hardware
//
#include <hip/hip_runtime.h>
#include <stdint.h>

// ---------------------------------------------------------------------------
// Seq2seq (biLSTM encoder + attention LSTM decoder) for MI455X / gfx1250.
// All GEMMs use v_wmma_f32_16x16x32_f16 (f16 inputs, fp32 accumulate).
// LSTM cells are fully fused: one wave owns a 16x16 (batch x hidden) patch and
// carries 4 accumulators (i,f,g,o gate tiles). Inner loops are software-
// pipelined one stage deep so fragment loads for iteration kt+1 overlap the
// WMMAs of iteration kt. Waves within a workgroup share the SAME weight
// fragment addresses (block = N-strip, waves = M-tiles) so the dominant
// weight stream hits the WGP cache instead of being replicated into L2.
// Weights are converted to f16 once per call and stay L2-resident (~25MB).
// ---------------------------------------------------------------------------

typedef __attribute__((ext_vector_type(8)))  _Float16 v8h;
typedef __attribute__((ext_vector_type(16))) _Float16 v16h;
typedef __attribute__((ext_vector_type(8)))  float    v8f;

#define Bsz 128
#define Tsz 256
#define Hsz 512
#define Ssz 32

__device__ __forceinline__ float sigmoid_f(float x) { return 1.0f / (1.0f + expf(-x)); }

// A fragment: 16x32 f16, lane<16 holds row (m0+lane), K = {0..7, 16..23};
// lane>=16 holds row (m0+lane-16), K = {8..15, 24..31}. Two b128 loads/lane.
__device__ __forceinline__ v16h load_fragA(const _Float16* base, int lda,
                                           int row0, int kt, int lane) {
  int r = lane & 15, hf = lane >> 4;
  const _Float16* p = base + (size_t)(row0 + r) * lda + kt * 32 + hf * 8;
  v8h lo = *(const v8h*)p;
  v8h hi = *(const v8h*)(p + 16);
  return __builtin_shufflevector(lo, hi, 0,1,2,3,4,5,6,7,8,9,10,11,12,13,14,15);
}

// B fragment from row-major W (N x K): lane<16 holds col (n0+lane), K = {0..7, 16..23};
// lane>=16 holds col (n0+lane-16), K = {8..15, 24..31}. Two b128 loads/lane.
__device__ __forceinline__ v16h load_fragB(const _Float16* base, int ldb,
                                           int col0, int kt, int lane) {
  int c = lane & 15, hf = lane >> 4;
  const _Float16* p = base + (size_t)(col0 + c) * ldb + kt * 32 + hf * 8;
  v8h lo = *(const v8h*)p;
  v8h hi = *(const v8h*)(p + 16);
  return __builtin_shufflevector(lo, hi, 0,1,2,3,4,5,6,7,8,9,10,11,12,13,14,15);
}

// One K-segment of the 4-tile GEMM, software-pipelined 1 stage deep.
// n0..n3 are the 4 B column-tile bases sharing the same A fragment.
__device__ __forceinline__ void gemm_segment4(
    const _Float16* A, int lda, int nkt, int row0,
    const _Float16* W, int ldw, int n0, int n1, int n2, int n3, int lane,
    v8f& acc0, v8f& acc1, v8f& acc2, v8f& acc3) {
  if (nkt <= 0) return;
  v16h a  = load_fragA(A, lda, row0, 0, lane);
  v16h b0 = load_fragB(W, ldw, n0, 0, lane);
  v16h b1 = load_fragB(W, ldw, n1, 0, lane);
  v16h b2 = load_fragB(W, ldw, n2, 0, lane);
  v16h b3 = load_fragB(W, ldw, n3, 0, lane);
  for (int kt = 0; kt + 1 < nkt; ++kt) {
    // prefetch next iteration's fragments before consuming current ones
    v16h an = load_fragA(A, lda, row0, kt + 1, lane);
    v16h c0 = load_fragB(W, ldw, n0, kt + 1, lane);
    v16h c1 = load_fragB(W, ldw, n1, kt + 1, lane);
    v16h c2 = load_fragB(W, ldw, n2, kt + 1, lane);
    v16h c3 = load_fragB(W, ldw, n3, kt + 1, lane);
    acc0 = __builtin_amdgcn_wmma_f32_16x16x32_f16(false, a, false, b0, (short)0, acc0, false, false);
    acc1 = __builtin_amdgcn_wmma_f32_16x16x32_f16(false, a, false, b1, (short)0, acc1, false, false);
    acc2 = __builtin_amdgcn_wmma_f32_16x16x32_f16(false, a, false, b2, (short)0, acc2, false, false);
    acc3 = __builtin_amdgcn_wmma_f32_16x16x32_f16(false, a, false, b3, (short)0, acc3, false, false);
    a = an; b0 = c0; b1 = c1; b2 = c2; b3 = c3;
  }
  acc0 = __builtin_amdgcn_wmma_f32_16x16x32_f16(false, a, false, b0, (short)0, acc0, false, false);
  acc1 = __builtin_amdgcn_wmma_f32_16x16x32_f16(false, a, false, b1, (short)0, acc1, false, false);
  acc2 = __builtin_amdgcn_wmma_f32_16x16x32_f16(false, a, false, b2, (short)0, acc2, false, false);
  acc3 = __builtin_amdgcn_wmma_f32_16x16x32_f16(false, a, false, b3, (short)0, acc3, false, false);
}

// ------------------------- fused LSTM cell kernel --------------------------
struct CellDir {
  const _Float16* A0; int lda0; int kt0;   // first K segment of activations
  const _Float16* A1; int lda1; int kt1;   // second K segment (recurrent h)
  const _Float16* W;  int ldw;             // combined weights (2048 x 32*(kt0+kt1))
  const float* bias;                       // 4H fp32
  float* c;                                // (128,512) fp32 cell state, in-place
  _Float16* h1; int ldh1;                  // f16 h output (primary)
  _Float16* h2; int ldh2;                  // f16 h output (secondary, may be null)
};

// grid: x = 32 (hidden tile j), z = ndir; block: 256 threads = 8 waves = 8
// batch tiles. All 8 waves in a block stream identical weight fragments.
__global__ __launch_bounds__(256) void lstm_cell_fused(CellDir dF, CellDir dB) {
  const CellDir d = (blockIdx.z == 0) ? dF : dB;
  const int lane = threadIdx.x & 31;
  const int m = threadIdx.x >> 5;   // batch tile 0..7 (per-wave within block)
  const int j = blockIdx.x;         // hidden tile 0..31 (shared by block)
  const int n0 = (0 * 32 + j) * 16, n1 = (1 * 32 + j) * 16;
  const int n2 = (2 * 32 + j) * 16, n3 = (3 * 32 + j) * 16;

  const int col   = j * 16 + (lane & 15);
  const int rbase = m * 16 + (lane >> 4) * 8;
  // warm the fp32 cell-state lines while the GEMM runs
#pragma unroll
  for (int v = 0; v < 8; ++v)
    __builtin_prefetch(&d.c[(rbase + v) * Hsz + col], 1, 3);

  v8f acc0 = {}, acc1 = {}, acc2 = {}, acc3 = {};
  gemm_segment4(d.A0, d.lda0, d.kt0, m * 16, d.W, d.ldw, n0, n1, n2, n3, lane,
                acc0, acc1, acc2, acc3);
  // second segment starts at K-tile kt0 within W
  gemm_segment4(d.A1, d.lda1, d.kt1, m * 16, d.W + d.kt0 * 32, d.ldw,
                n0, n1, n2, n3, lane, acc0, acc1, acc2, acc3);

  const float bi = d.bias[col];
  const float bf = d.bias[512 + col];
  const float bg = d.bias[1024 + col];
  const float bo = d.bias[1536 + col];
#pragma unroll
  for (int v = 0; v < 8; ++v) {
    float iv = sigmoid_f(acc0[v] + bi);
    float fv = sigmoid_f(acc1[v] + bf);
    float gv = tanhf(acc2[v] + bg);
    float ov = sigmoid_f(acc3[v] + bo);
    int r = rbase + v;
    int cidx = r * Hsz + col;
    float cv = fv * d.c[cidx] + iv * gv;
    d.c[cidx] = cv;
    _Float16 hv = (_Float16)(ov * tanhf(cv));
    d.h1[(size_t)r * d.ldh1 + col] = hv;
    if (d.h2) d.h2[(size_t)r * d.ldh2 + col] = hv;
  }
}

// ------------------------- generic WMMA GEMM (N blocked by 64) -------------
// C(M,N) = act( A(M,K f16) @ W(N,K f16)^T + bias ). Each wave computes a
// 16x64 strip (4 consecutive N tiles) reusing its A fragment 4x. A block's
// 8 waves cover 8 consecutive M tiles of the SAME N strip, sharing the weight
// stream. Requires M % 128 == 0 and N % 64 == 0 (true at all call sites).
__global__ __launch_bounds__(256) void wmma_gemm(
    const _Float16* A, int lda, const _Float16* W, int ldw,
    const float* bias, int M, int N, int ktiles,
    float* C, int ldc, _Float16* Hout, int ldh, int act) {
  const int lane = threadIdx.x & 31;
  const int nstrips = N >> 6;                       // strips of 64 columns
  const int mg = blockIdx.x / nstrips;              // group of 8 M tiles
  const int ns = blockIdx.x % nstrips;
  const int mt = mg * 8 + (threadIdx.x >> 5);
  const int n0 = ns * 64, n1 = n0 + 16, n2 = n0 + 32, n3 = n0 + 48;

  v8f acc0 = {}, acc1 = {}, acc2 = {}, acc3 = {};
  gemm_segment4(A, lda, ktiles, mt * 16, W, ldw, n0, n1, n2, n3, lane,
                acc0, acc1, acc2, acc3);

  const int colb  = lane & 15;
  const int rbase = mt * 16 + (lane >> 4) * 8;
#pragma unroll
  for (int g = 0; g < 4; ++g) {
    const v8f acc = (g == 0) ? acc0 : (g == 1) ? acc1 : (g == 2) ? acc2 : acc3;
    const int col = n0 + g * 16 + colb;
    const float bv = bias ? bias[col] : 0.0f;
#pragma unroll
    for (int v = 0; v < 8; ++v) {
      float xv = acc[v] + bv;
      if (act) xv = tanhf(xv);
      if (C)    C[(size_t)(rbase + v) * ldc + col] = xv;
      if (Hout) Hout[(size_t)(rbase + v) * ldh + col] = (_Float16)xv;
    }
  }
}

// ------------------------- attention kernels -------------------------------
// score[b,t] = sum_a Va[a] * tanh(Wa_vals[b*T+t, a] + qUa[b, a])
__global__ __launch_bounds__(256) void attn_score(const float* WaV, const float* qUa,
                                                  const float* Va, float* score) {
  const int lane = threadIdx.x & 31;
  const int row = ((blockIdx.x << 8) + threadIdx.x) >> 5;  // b*T + t
  const int b = row >> 8;
  const float* wv = WaV + (size_t)row * Hsz;
  const float* qa = qUa + (size_t)b * Hsz;
  float s = 0.0f;
  for (int a = lane; a < Hsz; a += 32) s += Va[a] * tanhf(wv[a] + qa[a]);
#pragma unroll
  for (int o = 16; o >= 1; o >>= 1) s += __shfl_xor(s, o, 32);
  if (lane == 0) score[row] = s;
}

__global__ __launch_bounds__(256) void softmax_t(const float* score, float* wts) {
  __shared__ float red[256];
  const int b = blockIdx.x, t = threadIdx.x;
  float v = score[b * Tsz + t];
  red[t] = v; __syncthreads();
  for (int s = 128; s > 0; s >>= 1) { if (t < s) red[t] = fmaxf(red[t], red[t + s]); __syncthreads(); }
  float mx = red[0]; __syncthreads();
  float e = expf(v - mx);
  red[t] = e; __syncthreads();
  for (int s = 128; s > 0; s >>= 1) { if (t < s) red[t] += red[t + s]; __syncthreads(); }
  wts[b * Tsz + t] = e / red[0];
}

// z[b,c] = sum_t wts[b,t] * enc_out[b,t,c]   (c < 1024), written as f16
__global__ __launch_bounds__(256) void attn_context(const float* wts, const _Float16* enc,
                                                    _Float16* dst, int ldd) {
  const int idx = (blockIdx.x << 8) + threadIdx.x;  // b*1024 + c
  const int b = idx >> 10, c = idx & 1023;
  const _Float16* e = enc + (size_t)b * Tsz * 1024 + c;
  const float* w = wts + b * Tsz;
  float s = 0.0f;
  for (int t = 0; t < Tsz; ++t) s += w[t] * (float)e[(size_t)t * 1024];
  dst[(size_t)b * ldd + c] = (_Float16)s;
}

// yn[b,o] = h1[b,:] . fcW[o,:] + fcb[o]; write fp32 to out and f16 to next input
__global__ __launch_bounds__(256) void fc_step(const _Float16* h, int ldh,
                                               const float* fcW, const float* fcb,
                                               float* out, int s, _Float16* ynext, int ldd) {
  const int lane = threadIdx.x & 31;
  const int wave = ((blockIdx.x << 8) + threadIdx.x) >> 5;  // 0..255 = b*2+o
  const int b = wave >> 1, o = wave & 1;
  const _Float16* hp = h + (size_t)b * ldh;
  const float* w = fcW + o * Hsz;
  float acc = 0.0f;
  for (int k = lane; k < Hsz; k += 32) acc += (float)hp[k] * w[k];
#pragma unroll
  for (int m = 16; m >= 1; m >>= 1) acc += __shfl_xor(acc, m, 32);
  if (lane == 0) {
    float y = acc + fcb[o];
    out[b * (Ssz * 2) + s * 2 + o] = y;
    ynext[(size_t)b * ldd + o] = (_Float16)y;
  }
}

// ------------------------- utility kernels ---------------------------------
__global__ __launch_bounds__(256) void zero_fill(uint32_t* p, int n) {
  int i = (blockIdx.x << 8) + threadIdx.x;
  if (i < n) p[i] = 0;
}

__global__ __launch_bounds__(256) void cvt_f16(const float* src, int srcLd,
                                               _Float16* dst, int dstLd, int rows, int cols) {
  int idx = (blockIdx.x << 8) + threadIdx.x;
  if (idx >= rows * cols) return;
  int r = idx / cols, c = idx - r * cols;
  dst[(size_t)r * dstLd + c] = (_Float16)src[(size_t)r * srcLd + c];
}

// x (B,T,8) fp32 -> x16 (T,B,32) f16 (cols 8..31 stay zero)
__global__ __launch_bounds__(256) void build_x16(const float* x, _Float16* x16) {
  int idx = (blockIdx.x << 8) + threadIdx.x;  // b*2048 + t*8 + k, n = 262144
  if (idx >= Bsz * Tsz * 8) return;
  int k = idx & 7, t = (idx >> 3) & 255, b = idx >> 11;
  x16[((size_t)t * Bsz + b) * 32 + k] = (_Float16)x[idx];
}

// y0 = x[:, T-1, :2] -> f16 into D0 cols 0..1
__global__ __launch_bounds__(256) void init_y0(const float* x, _Float16* D0) {
  int tid = threadIdx.x;
  if (tid < 256) {
    int b = tid >> 1, o = tid & 1;
    D0[(size_t)b * 1568 + o] = (_Float16)x[(size_t)b * Tsz * 8 + (Tsz - 1) * 8 + o];
  }
}

// ---------------------------------------------------------------------------
extern "C" void kernel_launch(void* const* d_in, const int* in_sizes, int n_in,
                              void* d_out, int out_size, void* d_ws, size_t ws_size,
                              hipStream_t stream) {
  (void)in_sizes; (void)n_in; (void)out_size; (void)ws_size;
  const float* x     = (const float*)d_in[0];
  const float* eWih0 = (const float*)d_in[1];
  const float* eWhh0 = (const float*)d_in[2];
  const float* eb0   = (const float*)d_in[3];
  const float* eWih1 = (const float*)d_in[4];
  const float* eWhh1 = (const float*)d_in[5];
  const float* eb1   = (const float*)d_in[6];
  const float* dWih0 = (const float*)d_in[7];
  const float* dWhh0 = (const float*)d_in[8];
  const float* db0   = (const float*)d_in[9];
  const float* dWih1 = (const float*)d_in[10];
  const float* dWhh1 = (const float*)d_in[11];
  const float* db1   = (const float*)d_in[12];
  const float* Wa    = (const float*)d_in[13];
  const float* Ua    = (const float*)d_in[14];
  const float* Va    = (const float*)d_in[15];
  const float* projW = (const float*)d_in[16];
  const float* projb = (const float*)d_in[17];
  const float* fcW   = (const float*)d_in[18];
  const float* fcb   = (const float*)d_in[19];
  float* out = (float*)d_out;

  size_t off = 0;
  auto carve = [&](size_t bytes) -> void* {
    off = (off + 255) & ~(size_t)255;
    void* p = (char*)d_ws + off;
    off += bytes;
    return p;
  };

  // ---- zero-initialized region (zeroed every call) ----
  _Float16* zbuf = (_Float16*)carve((size_t)Bsz * 512 * 2);
  float* c0f = (float*)carve((size_t)Bsz * Hsz * 4);
  float* c0b = (float*)carve((size_t)Bsz * Hsz * 4);
  float* c1f = (float*)carve((size_t)Bsz * Hsz * 4);
  float* c1b = (float*)carve((size_t)Bsz * Hsz * 4);
  _Float16* D[2] = { (_Float16*)carve((size_t)Bsz * 1568 * 2),
                     (_Float16*)carve((size_t)Bsz * 1568 * 2) };
  _Float16* Q[2] = { (_Float16*)carve((size_t)Bsz * 1024 * 2),
                     (_Float16*)carve((size_t)Bsz * 1024 * 2) };
  _Float16* x16 = (_Float16*)carve((size_t)Tsz * Bsz * 32 * 2);
  _Float16* We0[2] = { (_Float16*)carve((size_t)2048 * 544 * 2),
                       (_Float16*)carve((size_t)2048 * 544 * 2) };
  _Float16* Wd0 = (_Float16*)carve((size_t)2048 * 1568 * 2);
  size_t zero_bytes = (off + 255) & ~(size_t)255;

  // ---- remaining scratch (fully overwritten, no zeroing needed) ----
  _Float16* We1[2] = { (_Float16*)carve((size_t)2048 * 1536 * 2),
                       (_Float16*)carve((size_t)2048 * 1536 * 2) };
  _Float16* Wd1  = (_Float16*)carve((size_t)2048 * 1024 * 2);
  _Float16* Wa16 = (_Float16*)carve((size_t)512 * 1024 * 2);
  _Float16* Ua16 = (_Float16*)carve((size_t)512 * 512 * 2);
  _Float16* Pr16 = (_Float16*)carve((size_t)512 * 512 * 2);
  _Float16* l1in   = (_Float16*)carve((size_t)Tsz * Bsz * 1024 * 2);  // layer0 out (T,B,2H)
  _Float16* encout = (_Float16*)carve((size_t)Bsz * Tsz * 1024 * 2);  // layer1 out (B,T,2H)
  float* WaV    = (float*)carve((size_t)Bsz * Tsz * Hsz * 4);          // (B*T, 512)
  float* scoreB = (float*)carve((size_t)Bsz * Tsz * 4);
  float* wtsB   = (float*)carve((size_t)Bsz * Tsz * 4);
  float* qUaB   = (float*)carve((size_t)Bsz * Hsz * 4);

  // ---- zero + weight conversion ----
  {
    int n = (int)(zero_bytes / 4);
    zero_fill<<<(n + 255) / 256, 256, 0, stream>>>((uint32_t*)d_ws, n);
  }
  auto cvt = [&](const float* src, int srcLd, _Float16* dst, int dstLd, int rows, int cols) {
    int n = rows * cols;
    cvt_f16<<<(n + 255) / 256, 256, 0, stream>>>(src, srcLd, dst, dstLd, rows, cols);
  };
  build_x16<<<(Bsz * Tsz * 8 + 255) / 256, 256, 0, stream>>>(x, x16);
  for (int d = 0; d < 2; ++d) {
    cvt(eWih0 + (size_t)d * 2048 * 8,    8,    We0[d],        544, 2048, 8);
    cvt(eWhh0 + (size_t)d * 2048 * 512,  512,  We0[d] + 32,   544, 2048, 512);
    cvt(eWih1 + (size_t)d * 2048 * 1024, 1024, We1[d],        1536, 2048, 1024);
    cvt(eWhh1 + (size_t)d * 2048 * 512,  512,  We1[d] + 1024, 1536, 2048, 512);
  }
  cvt(dWih0, 1026, Wd0,        1568, 2048, 1026);
  cvt(dWhh0, 512,  Wd0 + 1056, 1568, 2048, 512);
  cvt(dWih1, 512,  Wd1,        1024, 2048, 512);
  cvt(dWhh1, 512,  Wd1 + 512,  1024, 2048, 512);
  cvt(Wa,    1024, Wa16, 1024, 512, 1024);
  cvt(Ua,    512,  Ua16, 512,  512, 512);
  cvt(projW, 512,  Pr16, 512,  512, 512);

  auto cell = [&](const CellDir& a, const CellDir& b, int ndir) {
    lstm_cell_fused<<<dim3(32, 1, ndir), 256, 0, stream>>>(a, b);
  };
  auto gemm = [&](const _Float16* A, int lda, const _Float16* W, int ldw, const float* bias,
                  int M, int N, int ktiles, float* C, int ldc, _Float16* Hout, int ldh, int act) {
    int blocks = (M / 128) * (N / 64);   // 8 waves/block: 8 M-tiles x one N-strip
    wmma_gemm<<<blocks, 256, 0, stream>>>(A, lda, W, ldw, bias, M, N, ktiles,
                                          C, ldc, Hout, ldh, act);
  };

  // ---- encoder layer 0 (fwd + bwd per launch) ----
  for (int t = 0; t < Tsz; ++t) {
    int tb = Tsz - 1 - t;
    CellDir f{}, b{};
    f.A0 = x16 + (size_t)t * Bsz * 32; f.lda0 = 32; f.kt0 = 1;
    f.A1 = (t == 0) ? zbuf : l1in + (size_t)(t - 1) * Bsz * 1024;
    f.lda1 = (t == 0) ? 512 : 1024; f.kt1 = 16;
    f.W = We0[0]; f.ldw = 544; f.bias = eb0; f.c = c0f;
    f.h1 = l1in + (size_t)t * Bsz * 1024; f.ldh1 = 1024; f.h2 = nullptr; f.ldh2 = 0;

    b.A0 = x16 + (size_t)tb * Bsz * 32; b.lda0 = 32; b.kt0 = 1;
    b.A1 = (t == 0) ? zbuf : l1in + (size_t)(tb + 1) * Bsz * 1024 + 512;
    b.lda1 = (t == 0) ? 512 : 1024; b.kt1 = 16;
    b.W = We0[1]; b.ldw = 544; b.bias = eb0 + 2048; b.c = c0b;
    b.h1 = l1in + (size_t)tb * Bsz * 1024 + 512; b.ldh1 = 1024; b.h2 = nullptr; b.ldh2 = 0;
    cell(f, b, 2);
  }

  // ---- encoder layer 1 (fwd + bwd per launch) ----
  const int ldEnc = Tsz * 1024;  // row stride over batch in enc_out (B,T,2H)
  for (int t = 0; t < Tsz; ++t) {
    int tb = Tsz - 1 - t;
    CellDir f{}, b{};
    f.A0 = l1in + (size_t)t * Bsz * 1024; f.lda0 = 1024; f.kt0 = 32;
    f.A1 = (t == 0) ? zbuf : encout + (size_t)(t - 1) * 1024;
    f.lda1 = (t == 0) ? 512 : ldEnc; f.kt1 = 16;
    f.W = We1[0]; f.ldw = 1536; f.bias = eb1; f.c = c1f;
    f.h1 = encout + (size_t)t * 1024; f.ldh1 = ldEnc; f.h2 = nullptr; f.ldh2 = 0;

    b.A0 = l1in + (size_t)tb * Bsz * 1024; b.lda0 = 1024; b.kt0 = 32;
    b.A1 = (t == 0) ? zbuf : encout + (size_t)(tb + 1) * 1024 + 512;
    b.lda1 = (t == 0) ? 512 : ldEnc; b.kt1 = 16;
    b.W = We1[1]; b.ldw = 1536; b.bias = eb1 + 2048; b.c = c1b;
    b.h1 = encout + (size_t)tb * 1024 + 512; b.ldh1 = ldEnc; b.h2 = nullptr; b.ldh2 = 0;
    cell(f, b, 2);
  }

  // ---- attention precompute + decoder init ----
  gemm(encout, 1024, Wa16, 1024, nullptr, Bsz * Tsz, 512, 32, WaV, 512, nullptr, 0, 0);
  // dh[0] = tanh(hf0 @ projW^T + projb) -> decoder L0 h_init (D[0] cols 1056..)
  gemm(l1in + (size_t)(Tsz - 1) * Bsz * 1024, 1024, Pr16, 512, projb,
       Bsz, 512, 16, nullptr, 0, D[0] + 1056, 1568, 1);
  // dh[1] = tanh(hb0 @ projW^T + projb) -> decoder L1 h_init (Q[0] cols 512..)
  gemm(l1in + 512, 1024, Pr16, 512, projb,
       Bsz, 512, 16, nullptr, 0, Q[0] + 512, 1024, 1);
  init_y0<<<1, 256, 0, stream>>>(x, D[0]);
  // dc: decoder reuses c0f (=cf0) for layer0 and c0b (=cb0) for layer1 in place.

  // ---- decoder loop ----
  for (int s = 0; s < Ssz; ++s) {
    int p = s & 1, pn = p ^ 1;
    // q = h1(prev) lives at Q[p] cols 512..1023
    gemm(Q[p] + 512, 1024, Ua16, 512, nullptr, Bsz, 512, 16, qUaB, 512, nullptr, 0, 0);
    attn_score<<<(Bsz * Tsz) / 8, 256, 0, stream>>>(WaV, qUaB, Va, scoreB);
    softmax_t<<<Bsz, 256, 0, stream>>>(scoreB, wtsB);
    attn_context<<<(Bsz * 1024) / 256, 256, 0, stream>>>(wtsB, encout, D[p] + 2, 1568);

    CellDir c0{};
    c0.A0 = D[p]; c0.lda0 = 1568; c0.kt0 = 49;          // [y(2) z(1024) pad(30) h0(512)]
    c0.A1 = zbuf; c0.lda1 = 512;  c0.kt1 = 0;
    c0.W = Wd0; c0.ldw = 1568; c0.bias = db0; c0.c = c0f;
    c0.h1 = Q[p];        c0.ldh1 = 1024;                 // h0n -> cell1 input
    c0.h2 = D[pn] + 1056; c0.ldh2 = 1568;                // h0n -> next step recurrent
    cell(c0, c0, 1);

    CellDir c1{};
    c1.A0 = Q[p]; c1.lda0 = 1024; c1.kt0 = 32;           // [h0n(512) h1prev(512)]
    c1.A1 = zbuf; c1.lda1 = 512;  c1.kt1 = 0;
    c1.W = Wd1; c1.ldw = 1024; c1.bias = db1; c1.c = c0b;
    c1.h1 = Q[pn] + 512; c1.ldh1 = 1024;                 // h1n -> next step
    c1.h2 = nullptr; c1.ldh2 = 0;
    cell(c1, c1, 1);

    fc_step<<<32, 256, 0, stream>>>(Q[pn] + 512, 1024, fcW, fcb, out, s, D[pn], 1568);
  }
}
